// DeepseekV32SparseAttention_7327214206984
// MI455X (gfx1250) — matrix operand order repaired and checked by reading, not compiled, not run on hardware
//
#include <hip/hip_runtime.h>
#include <hip/hip_bf16.h>

// ---------------------------------------------------------------------------
// Problem constants (from reference)
// ---------------------------------------------------------------------------
constexpr int SEQ  = 2048;
constexpr int HID  = 2048;
constexpr int NHD  = 16;
constexpr int QLRk = 1536;
constexpr int KVLRk= 512;
constexpr int DRr  = 64;
constexpr int DNn  = 128;
constexpr int DVv  = 128;
constexpr int DQKk = DNn + DRr;      // 192
constexpr float EPSf = 1e-6f;

typedef __attribute__((ext_vector_type(16))) __bf16 v16bf;
typedef __attribute__((ext_vector_type(8)))  __bf16 v8bf;
typedef __attribute__((ext_vector_type(8)))  float  v8f;
typedef int vi4 __attribute__((vector_size(16)));   // matches builtin V4i param

// ---------------------------------------------------------------------------
// CDNA5 async global->LDS path (builtins confirmed present by round-3 probe).
// Signature: (V4i addrspace(1)*, V4i addrspace(3)*, imm offset, imm cpol)
// ---------------------------------------------------------------------------
#if defined(__has_builtin)
#  if __has_builtin(__builtin_amdgcn_global_load_async_to_lds_b128) && \
      __has_builtin(__builtin_amdgcn_s_wait_asynccnt)
#    define USE_ASYNC_LDS 1
#  endif
#endif
#ifndef USE_ASYNC_LDS
#  define USE_ASYNC_LDS 0
#endif

#define ASG __attribute__((address_space(1)))
#define ASL __attribute__((address_space(3)))

__device__ inline void copy16_g2l(const __bf16* __restrict__ gsrc, __bf16* ldst) {
#if USE_ASYNC_LDS
  __builtin_amdgcn_global_load_async_to_lds_b128(
      (ASG vi4*)gsrc, (ASL vi4*)ldst, 0, 0);
#else
  *(uint4*)ldst = *(const uint4*)gsrc;
#endif
}

__device__ inline void async_fence() {
#if USE_ASYNC_LDS
  __builtin_amdgcn_s_wait_asynccnt(0);
#endif
}

// ---------------------------------------------------------------------------
// WMMA helpers (gfx1250: V_WMMA_F32_16X16X32_BF16, wave32)
// ---------------------------------------------------------------------------
__device__ inline v8f wmma_bf16(v16bf a, v16bf b, v8f c) {
  return __builtin_amdgcn_wmma_f32_16x16x32_bf16(
      false, a, false, b, (short)0, c, false, false);
}

// A fragment: 16x32 (MxK) tile, row-major in LDS, stride lda (multiple of 8).
__device__ inline v16bf frag_a(const __bf16* base, int lda) {
  const int lane = threadIdx.x & 31;
  const int half = lane >> 4, r = lane & 15;
  const __bf16* p = base + r * lda + half * 8;
  const v8bf lo = *(const v8bf*)(p);
  const v8bf hi = *(const v8bf*)(p + 16);
  return __builtin_shufflevector(lo, hi, 0,1,2,3,4,5,6,7,8,9,10,11,12,13,14,15);
}

// B fragment: 32x16 (KxN) tile stored K-contiguous per column (stride ldk).
__device__ inline v16bf frag_bk(const __bf16* base, int ldk) {
  const int lane = threadIdx.x & 31;
  const int col = lane & 15, kb0 = (lane >> 4) * 8;
  const __bf16* p = base + col * ldk + kb0;
  const v8bf lo = *(const v8bf*)(p);
  const v8bf hi = *(const v8bf*)(p + 16);
  return __builtin_shufflevector(lo, hi, 0,1,2,3,4,5,6,7,8,9,10,11,12,13,14,15);
}

// ---------------------------------------------------------------------------
// NT GEMM: C[M,N] = A[M,K] * Bt[N,K]^T, bf16 in, fp32 accum.
// 256 threads = 8 waves; tile 128x128x64; wave owns 32 rows x 64 cols.
// ---------------------------------------------------------------------------
constexpr int TM = 128, TN = 128, TK = 64, TKP = TK + 8;

template <typename OUT>
__global__ __launch_bounds__(256)
void gemm_nt_kernel(const __bf16* __restrict__ A, const __bf16* __restrict__ Bt,
                    OUT* __restrict__ C, int M, int N, int K) {
  __shared__ __bf16 As[TM][TKP];
  __shared__ __bf16 Bs[TN][TKP];

  const int tid  = threadIdx.x;
  const int bm   = blockIdx.y * TM;
  const int bn   = blockIdx.x * TN;
  const int wave = tid >> 5;
  const int wm   = wave >> 1;   // rows wm*32
  const int wn   = wave & 1;    // cols wn*64

  v8f acc[2][4];
#pragma unroll
  for (int m = 0; m < 2; ++m)
#pragma unroll
    for (int n = 0; n < 4; ++n) acc[m][n] = {};

  for (int k0 = 0; k0 < K; k0 += TK) {
#pragma unroll
    for (int i = 0; i < 4; ++i) {            // 1024 chunks of 8 bf16 (16B)
      const int id = tid + i * 256;
      const int r = id >> 3, c8 = (id & 7) * 8;
      copy16_g2l(&A[(size_t)(bm + r) * K + k0 + c8], &As[r][c8]);
      if (bn + r < N) {
        copy16_g2l(&Bt[(size_t)(bn + r) * K + k0 + c8], &Bs[r][c8]);
      } else {
        *(uint4*)&Bs[r][c8] = uint4{0u, 0u, 0u, 0u};
      }
    }
    // prefetch next k-tile rows (global_prefetch_b8)
    if (k0 + TK < K) {
      if (tid < TM) __builtin_prefetch(&A[(size_t)(bm + tid) * K + k0 + TK], 0, 3);
      if (tid < TN && bn + tid < N)
        __builtin_prefetch(&Bt[(size_t)(bn + tid) * K + k0 + TK], 0, 3);
    }
    async_fence();
    __syncthreads();

#pragma unroll
    for (int kk = 0; kk < TK; kk += 32) {
      v16bf af[2], bb[4];
#pragma unroll
      for (int m = 0; m < 2; ++m) af[m] = frag_a(&As[wm * 32 + m * 16][kk], TKP);
#pragma unroll
      for (int n = 0; n < 4; ++n) bb[n] = frag_bk(&Bs[wn * 64 + n * 16][kk], TKP);
#pragma unroll
      for (int m = 0; m < 2; ++m)
#pragma unroll
        for (int n = 0; n < 4; ++n)
          acc[m][n] = wmma_bf16(af[m], bb[n], acc[m][n]);
    }
    __syncthreads();
  }

  const int lane = tid & 31, half = lane >> 4, colL = lane & 15;
#pragma unroll
  for (int m = 0; m < 2; ++m)
#pragma unroll
    for (int n = 0; n < 4; ++n)
#pragma unroll
      for (int v = 0; v < 8; ++v) {
        const int gr = bm + wm * 32 + m * 16 + half * 8 + v;
        const int gc = bn + wn * 64 + n * 16 + colL;
        if (gc < N) C[(size_t)gr * N + gc] = (OUT)acc[m][n][v];
      }
}

// ---------------------------------------------------------------------------
// fp32 -> bf16 bulk convert (4 elements / thread)
// ---------------------------------------------------------------------------
__global__ void cvt_bf16_kernel(const float* __restrict__ x, __bf16* __restrict__ y,
                                size_t n4) {
  const size_t i = (size_t)blockIdx.x * 256 + threadIdx.x;
  if (i >= n4) return;
  const float4 v = ((const float4*)x)[i];
  union { __bf16 h[4]; uint2 u; } pk;
  pk.h[0] = (__bf16)v.x; pk.h[1] = (__bf16)v.y;
  pk.h[2] = (__bf16)v.z; pk.h[3] = (__bf16)v.w;
  *(uint2*)&y[4 * i] = pk.u;
}

// ---------------------------------------------------------------------------
// Transpose + convert: W[K][N] fp32 -> Wt[N][K] bf16 (tiled 32x32 via LDS)
// ---------------------------------------------------------------------------
__global__ __launch_bounds__(256)
void transcvt_kernel(const float* __restrict__ W, __bf16* __restrict__ Wt,
                     int K, int N) {
  __shared__ float t[32][33];
  const int bn = blockIdx.x * 32;
  const int bk = blockIdx.y * 32;
  const int tx = threadIdx.x & 31, ty = threadIdx.x >> 5;
#pragma unroll
  for (int i = 0; i < 32; i += 8) {
    const int k = bk + ty + i, n = bn + tx;
    t[ty + i][tx] = (k < K && n < N) ? W[(size_t)k * N + n] : 0.f;
  }
  __syncthreads();
#pragma unroll
  for (int i = 0; i < 32; i += 8) {
    const int n = bn + ty + i, k = bk + tx;
    if (n < N && k < K) Wt[(size_t)n * K + k] = (__bf16)t[tx][ty + i];
  }
}

// ---------------------------------------------------------------------------
// V transpose (bf16->bf16): Vt[h*DV + d][s] = kv_bf[s][h*(DN+DV) + DN + d]
// ---------------------------------------------------------------------------
__global__ __launch_bounds__(256)
void vtrans_kernel(const __bf16* __restrict__ kvb, __bf16* __restrict__ Vt) {
  __shared__ __bf16 t[32][40];
  const int h  = blockIdx.z;
  const int s0 = blockIdx.x * 32, d0 = blockIdx.y * 32;
  const int tx = threadIdx.x & 31, ty = threadIdx.x >> 5;
#pragma unroll
  for (int i = 0; i < 32; i += 8)
    t[ty + i][tx] =
        kvb[(size_t)(s0 + ty + i) * (NHD * (DNn + DVv)) + h * (DNn + DVv) + DNn + d0 + tx];
  __syncthreads();
#pragma unroll
  for (int i = 0; i < 32; i += 8)
    Vt[(size_t)(h * DVv + d0 + ty + i) * SEQ + s0 + tx] = t[tx][ty + i];
}

// ---------------------------------------------------------------------------
// RMSNorm (fp32 in, bf16 out)
// ---------------------------------------------------------------------------
__global__ __launch_bounds__(256)
void rmsnorm_bf16_kernel(const float* __restrict__ x, const float* __restrict__ w,
                         __bf16* __restrict__ y, int L, int strideIn, int strideOut) {
  __shared__ float red[256];
  const int row = blockIdx.x;
  const int tid = threadIdx.x;
  const float* xr = x + (size_t)row * strideIn;
  float s = 0.f;
  for (int i = tid; i < L; i += 256) { const float v = xr[i]; s += v * v; }
  red[tid] = s; __syncthreads();
  for (int k = 128; k > 0; k >>= 1) {
    if (tid < k) red[tid] += red[tid + k];
    __syncthreads();
  }
  const float inv = rsqrtf(red[0] / (float)L + EPSf);
  for (int i = tid; i < L; i += 256)
    y[(size_t)row * strideOut + i] = (__bf16)(xr[i] * inv * w[i]);
}

// ---------------------------------------------------------------------------
// RoPE (interleaved, from reference) + convert + fold in attention scale.
// ---------------------------------------------------------------------------
__global__ void rope_q_bf_kernel(const float* __restrict__ q, __bf16* __restrict__ qb,
                                 const float* __restrict__ cosb,
                                 const float* __restrict__ sinb, float scale) {
  const int idx = blockIdx.x * blockDim.x + threadIdx.x;   // s*NHD + h
  if (idx >= SEQ * NHD) return;
  const int s = idx / NHD, h = idx % NHD;
  const float* src = q + (size_t)s * (NHD * DQKk) + h * DQKk;
  __bf16* dst = qb + (size_t)s * (NHD * DQKk) + h * DQKk;
  const float* c  = cosb + (size_t)s * DRr;
  const float* sn = sinb + (size_t)s * DRr;
#pragma unroll
  for (int i = 0; i < DNn; ++i) dst[i] = (__bf16)(src[i] * scale);
  const float* v = src + DNn;
#pragma unroll
  for (int i = 0; i < 32; ++i) {
    const float a = v[2 * i], b = v[2 * i + 1];
    dst[DNn + i]      = (__bf16)((a * c[i]      - b * sn[i])      * scale);
    dst[DNn + 32 + i] = (__bf16)((b * c[32 + i] + a * sn[32 + i]) * scale);
  }
}

__global__ void rope_k_bf_kernel(const float* __restrict__ ckv, __bf16* __restrict__ krot,
                                 const float* __restrict__ cosb,
                                 const float* __restrict__ sinb) {
  const int s = blockIdx.x * blockDim.x + threadIdx.x;
  if (s >= SEQ) return;
  const float* v  = ckv + (size_t)s * (KVLRk + DRr) + KVLRk;
  const float* c  = cosb + (size_t)s * DRr;
  const float* sn = sinb + (size_t)s * DRr;
  __bf16* o = krot + (size_t)s * DRr;
#pragma unroll
  for (int i = 0; i < 32; ++i) {
    const float a = v[2 * i], b = v[2 * i + 1];
    o[i]      = (__bf16)(a * c[i]      - b * sn[i]);
    o[32 + i] = (__bf16)(b * c[32 + i] + a * sn[32 + i]);
  }
}

// ---------------------------------------------------------------------------
// Flash attention (causal). 64 q-rows x 64 kv per step, 8 waves.
// Q pre-scaled by 1/sqrt(192). All tile staging = contiguous 16B copies.
// ---------------------------------------------------------------------------
constexpr int AQ = 64, AK = 64;
constexpr int QKP = DQKk + 8;    // 200
constexpr int AKP = AK + 8;      // 72

__global__ __launch_bounds__(256)
void mla_attn_kernel(const __bf16* __restrict__ qb, const __bf16* __restrict__ kvb,
                     const __bf16* __restrict__ krot, const __bf16* __restrict__ Vt,
                     __bf16* __restrict__ attn) {
  __shared__ __bf16 Qs[AQ][QKP];
  __shared__ __bf16 Ks[AK][QKP];
  __shared__ __bf16 Vc[DVv][AKP];
  __shared__ float  sc[AQ][AKP];
  __shared__ __bf16 Ps[AQ][AKP];
  __shared__ float  row_m[AQ], row_l[AQ], row_scale[AQ];

  const int tid = threadIdx.x;
  const int qbk = blockIdx.x;
  const int h   = blockIdx.y;
  const int q0  = qbk * AQ;

  // Q tile: 64x192 = 1536 16B-chunks
#pragma unroll
  for (int i = 0; i < 6; ++i) {
    const int id = tid + i * 256;
    const int r = id / 24, c8 = (id % 24) * 8;
    copy16_g2l(&qb[(size_t)(q0 + r) * (NHD * DQKk) + h * DQKk + c8], &Qs[r][c8]);
  }
  if (tid < AQ) { row_m[tid] = -1e30f; row_l[tid] = 0.f; }

  const int lane = tid & 31, half = lane >> 4, colL = lane & 15;
  const int wave = tid >> 5, wm = wave >> 1, wn = wave & 1;

  v8f oacc[4];
#pragma unroll
  for (int n = 0; n < 4; ++n) oacc[n] = {};

  for (int j = 0; j <= qbk; ++j) {
    const int k0 = j * AK;
    async_fence();
    __syncthreads();   // prev P*V done before K/V/P/sc overwrite (also Q visible)

    // K tile: per key, dims 0..127 from kv, 128..191 from krot
#pragma unroll
    for (int i = 0; i < 6; ++i) {
      const int id = tid + i * 256;
      const int r = id / 24, g = id % 24;
      const size_t key = (size_t)(k0 + r);
      if (g < 16)
        copy16_g2l(&kvb[key * (NHD * (DNn + DVv)) + h * (DNn + DVv) + g * 8],
                   &Ks[r][g * 8]);
      else
        copy16_g2l(&krot[key * DRr + (g - 16) * 8], &Ks[r][g * 8]);
    }
    // V tile: dim-major from Vt
#pragma unroll
    for (int i = 0; i < 4; ++i) {
      const int id = tid + i * 256;
      const int d = id >> 3, k8 = (id & 7) * 8;
      copy16_g2l(&Vt[(size_t)(h * DVv + d) * SEQ + k0 + k8], &Vc[d][k8]);
    }
    // prefetch next kv block while this one computes
    if (j < qbk) {
      const int kn = k0 + AK;
      if (tid < AK)
        __builtin_prefetch(&kvb[(size_t)(kn + tid) * (NHD * (DNn + DVv)) + h * (DNn + DVv)], 0, 3);
      if (tid < DVv)
        __builtin_prefetch(&Vt[(size_t)(h * DVv + tid) * SEQ + kn], 0, 3);
    }
    async_fence();
    __syncthreads();

    // S = Q*K^T : wave -> rows wm*16, cols wn*32 (2 subtiles), d = 192
    v8f sacc[2]; sacc[0] = {}; sacc[1] = {};
#pragma unroll
    for (int kk = 0; kk < DQKk; kk += 32) {
      const v16bf aq = frag_a(&Qs[wm * 16][kk], QKP);
      const v16bf b0 = frag_bk(&Ks[wn * 32][kk], QKP);
      const v16bf b1 = frag_bk(&Ks[wn * 32 + 16][kk], QKP);
      sacc[0] = wmma_bf16(aq, b0, sacc[0]);
      sacc[1] = wmma_bf16(aq, b1, sacc[1]);
    }
#pragma unroll
    for (int n = 0; n < 2; ++n)
#pragma unroll
      for (int v = 0; v < 8; ++v)
        sc[wm * 16 + half * 8 + v][wn * 32 + n * 16 + colL] = sacc[n][v];
    __syncthreads();

    // Online softmax: one thread per query row (scores pre-scaled via Q).
    if (tid < AQ) {
      const int r = tid;
      const int grow = q0 + r;
      const float mprev = row_m[r];
      float mx = mprev;
      for (int c = 0; c < AK; ++c) {
        const float sv = (k0 + c <= grow) ? sc[r][c] : -1e30f;
        sc[r][c] = sv;
        mx = fmaxf(mx, sv);
      }
      float sum = 0.f;
      for (int c = 0; c < AK; ++c) {
        const float p = __expf(sc[r][c] - mx);
        Ps[r][c] = (__bf16)p;
        sum += p;
      }
      const float corr = __expf(mprev - mx);
      row_scale[r] = corr;
      row_m[r] = mx;
      row_l[r] = row_l[r] * corr + sum;
    }
    __syncthreads();

    // O = O*corr + P*V : wave -> rows wm*16, cols wn*64 (4 subtiles)
#pragma unroll
    for (int n = 0; n < 4; ++n)
#pragma unroll
      for (int v = 0; v < 8; ++v)
        oacc[n][v] *= row_scale[wm * 16 + half * 8 + v];
#pragma unroll
    for (int kk = 0; kk < AK; kk += 32) {
      const v16bf ap = frag_a(&Ps[wm * 16][kk], AKP);
#pragma unroll
      for (int n = 0; n < 4; ++n) {
        const v16bf bv = frag_bk(&Vc[wn * 64 + n * 16][kk], AKP);
        oacc[n] = wmma_bf16(ap, bv, oacc[n]);
      }
    }
  }
  __syncthreads();

  // Normalize and write bf16: attn[s, h*DV + c]
#pragma unroll
  for (int n = 0; n < 4; ++n)
#pragma unroll
    for (int v = 0; v < 8; ++v) {
      const int r = wm * 16 + half * 8 + v;
      const int c = wn * 64 + n * 16 + colL;
      attn[(size_t)(q0 + r) * (NHD * DVv) + h * DVv + c] =
          (__bf16)(oacc[n][v] / row_l[r]);
    }
}

// ---------------------------------------------------------------------------
// Launcher
// ---------------------------------------------------------------------------
extern "C" void kernel_launch(void* const* d_in, const int* in_sizes, int n_in,
                              void* d_out, int out_size, void* d_ws, size_t ws_size,
                              hipStream_t stream) {
  const float* hs    = (const float*)d_in[0];
  const float* cosb  = (const float*)d_in[1];
  const float* sinb  = (const float*)d_in[2];
  const float* wq_a  = (const float*)d_in[3];
  const float* q_ln  = (const float*)d_in[4];
  const float* wq_b  = (const float*)d_in[5];
  const float* wkv_a = (const float*)d_in[6];
  const float* kv_ln = (const float*)d_in[7];
  const float* wkv_b = (const float*)d_in[8];
  const float* wo    = (const float*)d_in[9];
  float* out = (float*)d_out;

  char* cur = (char*)d_ws;
  auto alloc_f = [&](size_t n) {
    float* r = (float*)cur; cur += ((n * 4 + 255) & ~(size_t)255); return r; };
  auto alloc_b = [&](size_t n) {
    __bf16* r = (__bf16*)cur; cur += ((n * 2 + 255) & ~(size_t)255); return r; };

  __bf16* hs_bf   = alloc_b((size_t)SEQ * HID);
  __bf16* wq_aT   = alloc_b((size_t)QLRk * HID);
  __bf16* wq_bT   = alloc_b((size_t)NHD * DQKk * QLRk);
  __bf16* wkv_aT  = alloc_b((size_t)(KVLRk + DRr) * HID);
  __bf16* wkv_bT  = alloc_b((size_t)NHD * (DNn + DVv) * KVLRk);
  __bf16* woT     = alloc_b((size_t)HID * NHD * DVv);
  float*  q_lat   = alloc_f((size_t)SEQ * QLRk);
  __bf16* qlat_bf = alloc_b((size_t)SEQ * QLRk);
  float*  qbuf    = alloc_f((size_t)SEQ * NHD * DQKk);
  __bf16* qbf     = alloc_b((size_t)SEQ * NHD * DQKk);
  float*  ckv     = alloc_f((size_t)SEQ * (KVLRk + DRr));
  __bf16* kvn_bf  = alloc_b((size_t)SEQ * KVLRk);
  __bf16* krot_bf = alloc_b((size_t)SEQ * DRr);
  __bf16* kv_bf   = alloc_b((size_t)SEQ * NHD * (DNn + DVv));
  __bf16* Vt      = alloc_b((size_t)NHD * DVv * SEQ);
  __bf16* attn_bf = alloc_b((size_t)SEQ * NHD * DVv);

  const dim3 blk(256);
  const float scale = rsqrtf((float)DQKk);

  // 0) one-time conversions / transposes (bf16 data path)
  cvt_bf16_kernel<<<(SEQ * HID / 4 + 255) / 256, blk, 0, stream>>>(hs, hs_bf,
                                                                   (size_t)SEQ * HID / 4);
  transcvt_kernel<<<dim3((QLRk + 31) / 32, (HID + 31) / 32), blk, 0, stream>>>(
      wq_a, wq_aT, HID, QLRk);
  transcvt_kernel<<<dim3((NHD * DQKk + 31) / 32, (QLRk + 31) / 32), blk, 0, stream>>>(
      wq_b, wq_bT, QLRk, NHD * DQKk);
  transcvt_kernel<<<dim3((KVLRk + DRr + 31) / 32, (HID + 31) / 32), blk, 0, stream>>>(
      wkv_a, wkv_aT, HID, KVLRk + DRr);
  transcvt_kernel<<<dim3((NHD * (DNn + DVv) + 31) / 32, (KVLRk + 31) / 32), blk, 0, stream>>>(
      wkv_b, wkv_bT, KVLRk, NHD * (DNn + DVv));
  transcvt_kernel<<<dim3((HID + 31) / 32, (NHD * DVv + 31) / 32), blk, 0, stream>>>(
      wo, woT, NHD * DVv, HID);

  // 1) q_lat = hs @ wq_a
  gemm_nt_kernel<float><<<dim3(QLRk / TN, SEQ / TM), blk, 0, stream>>>(
      hs_bf, wq_aT, q_lat, SEQ, QLRk, HID);
  // 2) qlat_bf = rmsnorm(q_lat)
  rmsnorm_bf16_kernel<<<SEQ, blk, 0, stream>>>(q_lat, q_ln, qlat_bf, QLRk, QLRk, QLRk);
  // 3) qbuf = qlat_bf @ wq_b
  gemm_nt_kernel<float><<<dim3((NHD * DQKk) / TN, SEQ / TM), blk, 0, stream>>>(
      qlat_bf, wq_bT, qbuf, SEQ, NHD * DQKk, QLRk);
  // 4) qbf = rope(qbuf) * scale (bf16)
  rope_q_bf_kernel<<<(SEQ * NHD + 255) / 256, blk, 0, stream>>>(qbuf, qbf, cosb, sinb, scale);
  // 5) ckv = hs @ wkv_a
  gemm_nt_kernel<float><<<dim3((KVLRk + DRr + TN - 1) / TN, SEQ / TM), blk, 0, stream>>>(
      hs_bf, wkv_aT, ckv, SEQ, KVLRk + DRr, HID);
  // 6) kvn_bf = rmsnorm(ckv[:, :512])
  rmsnorm_bf16_kernel<<<SEQ, blk, 0, stream>>>(ckv, kv_ln, kvn_bf, KVLRk, KVLRk + DRr, KVLRk);
  // 7) krot_bf = rope(ckv[:, 512:576])
  rope_k_bf_kernel<<<(SEQ + 255) / 256, blk, 0, stream>>>(ckv, krot_bf, cosb, sinb);
  // 8) kv_bf = kvn_bf @ wkv_b (bf16 out)
  gemm_nt_kernel<__bf16><<<dim3((NHD * (DNn + DVv)) / TN, SEQ / TM), blk, 0, stream>>>(
      kvn_bf, wkv_bT, kv_bf, SEQ, NHD * (DNn + DVv), KVLRk);
  // 9) Vt = transpose of V part of kv_bf
  vtrans_kernel<<<dim3(SEQ / 32, DVv / 32, NHD), blk, 0, stream>>>(kv_bf, Vt);
  // 10) attention
  mla_attn_kernel<<<dim3(SEQ / AQ, NHD), blk, 0, stream>>>(qbf, kv_bf, krot_bf, Vt, attn_bf);
  // 11) out = attn_bf @ wo
  gemm_nt_kernel<float><<<dim3(HID / TN, SEQ / TM), blk, 0, stream>>>(
      attn_bf, woT, out, SEQ, HID, NHD * DVv);
}
